// LSTMModel_14439680049454
// MI455X (gfx1250) — matrix operand order repaired and checked by reading, not compiled, not run on hardware
//
#include <hip/hip_runtime.h>
#include <math.h>

// ---------------------------------------------------------------------------
// LSTM (SEQ=4096, IN=HID=1024) for MI455X / gfx1250, wave32.
//   K1: embedding gather  -> bf16 X            [4096,1024]
//   K2: W_ih              -> bf16              [4096,1024]
//   K3: WMMA bf16 GEMM    xg = X @ W_ih^T + (b_ih+b_hh)   [4096,4096] f32
//       wave tile 16x64, double-buffered fragments (loads overlap WMMA)
//   K4: init h_buf[0]=h0, zero grid-barrier counter
//   K5: persistent 64-block LSTM scan; W_hh slice (256KB) pinned in LDS
//   K6: log_softmax over final h -> d_out [1,1024]
// ---------------------------------------------------------------------------

#define SEQ  4096
#define HID  1024
#define GATES (4 * HID)          // 4096

typedef __attribute__((ext_vector_type(16))) __bf16 v16bf;
typedef __attribute__((ext_vector_type(8)))  float  v8f;

union FragBF16 { v16bf v; uint4 q[2]; };

struct KFrags { FragBF16 a, b0, b1, b2, b3; };

__device__ __forceinline__ void load_frags(KFrags& f,
                                           const __bf16* __restrict__ arow,
                                           const __bf16* __restrict__ brow,
                                           int k0, int g) {
  // A 16x32 bf16: lane holds 16 contiguous K at k0+g*16 (matches B)
  f.a.q[0] = *(const uint4*)(arow + k0 + g * 16);
  f.a.q[1] = *(const uint4*)(arow + k0 + g * 16 + 8);
  // B 32x16 bf16: lane n holds 16 contiguous K at k0+g*16 (a W_ih row)
  const int kb = k0 + g * 16;
  f.b0.q[0] = *(const uint4*)(brow + 0 * 16 * HID + kb);
  f.b0.q[1] = *(const uint4*)(brow + 0 * 16 * HID + kb + 8);
  f.b1.q[0] = *(const uint4*)(brow + 1 * 16 * HID + kb);
  f.b1.q[1] = *(const uint4*)(brow + 1 * 16 * HID + kb + 8);
  f.b2.q[0] = *(const uint4*)(brow + 2 * 16 * HID + kb);
  f.b2.q[1] = *(const uint4*)(brow + 2 * 16 * HID + kb + 8);
  f.b3.q[0] = *(const uint4*)(brow + 3 * 16 * HID + kb);
  f.b3.q[1] = *(const uint4*)(brow + 3 * 16 * HID + kb + 8);
}

__device__ __forceinline__ unsigned short f32_to_bf16(float f) {
  unsigned int u = __float_as_uint(f);
  unsigned int r = u + 0x7FFFu + ((u >> 16) & 1u);   // round-to-nearest-even
  return (unsigned short)(r >> 16);
}

// ---- K1: gather embedding rows and convert to bf16 (4 elems/thread) --------
__global__ void embed_bf16_kernel(const int* __restrict__ tok,
                                  const float* __restrict__ emb,
                                  unsigned short* __restrict__ Xb) {
  int idx4 = (blockIdx.x * blockDim.x + threadIdx.x) * 4;   // [0, SEQ*HID)
  int s = idx4 >> 10, j = idx4 & 1023;
  float4 v = *(const float4*)(emb + (long long)tok[s] * HID + j);
  ushort4 o;
  o.x = f32_to_bf16(v.x); o.y = f32_to_bf16(v.y);
  o.z = f32_to_bf16(v.z); o.w = f32_to_bf16(v.w);
  *(ushort4*)(Xb + idx4) = o;
}

// ---- K2: convert W_ih to bf16 (4 elems/thread) -----------------------------
__global__ void wih_bf16_kernel(const float* __restrict__ W,
                                unsigned short* __restrict__ Wb) {
  int idx4 = (blockIdx.x * blockDim.x + threadIdx.x) * 4;   // [0, GATES*HID)
  float4 v = *(const float4*)(W + idx4);
  ushort4 o;
  o.x = f32_to_bf16(v.x); o.y = f32_to_bf16(v.y);
  o.z = f32_to_bf16(v.z); o.w = f32_to_bf16(v.w);
  *(ushort4*)(Wb + idx4) = o;
}

// ---- K3: xg = X @ W_ih^T + (b_ih + b_hh), WMMA bf16 -> f32 -----------------
// Block = 256 threads = 8 waves arranged 4(M) x 2(N).
// Wave tile = 16(M) x 64(N): one A fragment drives 4 WMMA per k-step.
// Fragments double-buffered: k-step i+1 loads issue before k-step i WMMAs,
// so waits are partial (overlap load stream with matrix pipe in-wave).
// Block tile = 64(M) x 128(N). Grid = (4096/64, 4096/128) = (64, 32).
__global__ void gemm_wmma_kernel(const __bf16* __restrict__ Xb,   // [SEQ][HID]
                                 const __bf16* __restrict__ Wb,   // [GATES][HID]
                                 const float* __restrict__ bih,
                                 const float* __restrict__ bhh,
                                 float* __restrict__ xg) {        // [SEQ][GATES]
  const int wave = threadIdx.x >> 5;
  const int lane = threadIdx.x & 31;
  const int m0 = blockIdx.x * 64 + (wave & 3) * 16;
  const int n0 = blockIdx.y * 128 + (wave >> 2) * 64;   // wave covers 64 cols

  const int l15 = lane & 15;          // row (A) / col (B/C)
  const int g   = lane >> 4;          // lane-group select

  const __bf16* arow = Xb + (long long)(m0 + l15) * HID;
  const __bf16* brow = Wb + (long long)(n0 + l15) * HID;  // + j*16*HID per tile

  v8f acc0 = {}, acc1 = {}, acc2 = {}, acc3 = {};
  KFrags f0, f1;
  load_frags(f0, arow, brow, 0, g);

  #pragma unroll 2
  for (int k0 = 0; k0 < HID; k0 += 64) {
    load_frags(f1, arow, brow, k0 + 32, g);   // in flight during f0 WMMAs
    acc0 = __builtin_amdgcn_wmma_f32_16x16x32_bf16(
        false, f0.a.v, false, f0.b0.v, (short)0, acc0, false, false);
    acc1 = __builtin_amdgcn_wmma_f32_16x16x32_bf16(
        false, f0.a.v, false, f0.b1.v, (short)0, acc1, false, false);
    acc2 = __builtin_amdgcn_wmma_f32_16x16x32_bf16(
        false, f0.a.v, false, f0.b2.v, (short)0, acc2, false, false);
    acc3 = __builtin_amdgcn_wmma_f32_16x16x32_bf16(
        false, f0.a.v, false, f0.b3.v, (short)0, acc3, false, false);

    if (k0 + 64 < HID)
      load_frags(f0, arow, brow, k0 + 64, g); // in flight during f1 WMMAs
    acc0 = __builtin_amdgcn_wmma_f32_16x16x32_bf16(
        false, f1.a.v, false, f1.b0.v, (short)0, acc0, false, false);
    acc1 = __builtin_amdgcn_wmma_f32_16x16x32_bf16(
        false, f1.a.v, false, f1.b1.v, (short)0, acc1, false, false);
    acc2 = __builtin_amdgcn_wmma_f32_16x16x32_bf16(
        false, f1.a.v, false, f1.b2.v, (short)0, acc2, false, false);
    acc3 = __builtin_amdgcn_wmma_f32_16x16x32_bf16(
        false, f1.a.v, false, f1.b3.v, (short)0, acc3, false, false);
  }

  // C/D layout: VGPR r, lanes 0-15 -> M=r, lanes 16-31 -> M=r+8; N = l15
  const int mrow = m0 + 8 * g;
  #pragma unroll
  for (int j = 0; j < 4; ++j) {
    const v8f& acc = (j == 0) ? acc0 : (j == 1) ? acc1 : (j == 2) ? acc2 : acc3;
    const int col = n0 + j * 16 + l15;
    const float bias = bih[col] + bhh[col];
    #pragma unroll
    for (int r = 0; r < 8; ++r) {
      xg[(long long)(mrow + r) * GATES + col] = acc[r] + bias;
    }
  }
}

// ---- K4: init h double-buffer and grid-barrier counter ---------------------
__global__ void init_state_kernel(const float* __restrict__ h0,
                                  float* __restrict__ h_buf,
                                  unsigned int* __restrict__ counter) {
  int t = threadIdx.x;
  if (t < HID) h_buf[t] = h0[t];       // h_buf[0][*] = h0
  if (t == 0) *counter = 0u;
}

// ---- K5: persistent LSTM scan ----------------------------------------------
// 64 blocks x 256 threads. Block b owns units [16b, 16b+16): 64 W_hh rows
// (i/f/g/o for those units) pinned in LDS for the whole scan (256 KB, legal
// within gfx1250's 320 KB WGP LDS; launched as dynamic LDS).
#define RBLOCKS 64
#define UNITS_PER_BLOCK 16
__global__ void lstm_scan_kernel(const float* __restrict__ Whh,  // [GATES][HID]
                                 const float* __restrict__ xg,   // [SEQ][GATES]
                                 const float* __restrict__ c0,
                                 float* __restrict__ h_buf,      // [2][HID]
                                 unsigned int* __restrict__ counter) {
  extern __shared__ char smem_raw[];
  float* w_lds    = (float*)smem_raw;            // [64][1024]  262144 B
  float* h_lds    = w_lds + 64 * HID;            // [1024]        4096 B
  float* part_lds = h_lds + HID;                 // [256]         1024 B
  float* gate_lds = part_lds + 256;              // [64]           256 B
  float* c_lds    = gate_lds + 64;               // [16]            64 B

  const int tid = threadIdx.x;
  const int blk = blockIdx.x;
  const int u_base = blk * UNITS_PER_BLOCK;

  // Load this block's 64 W_hh rows into LDS: row_local = gate*16 + u_local.
  for (int idx = tid; idx < 64 * HID; idx += 256) {
    int row_local = idx >> 10;
    int k = idx & 1023;
    int gate = row_local >> 4;
    int u = row_local & 15;
    int grow = gate * HID + u_base + u;
    w_lds[idx] = Whh[(long long)grow * HID + k];
  }
  if (tid < UNITS_PER_BLOCK) c_lds[tid] = c0[u_base + tid];
  __syncthreads();

  const int row_local = tid >> 2;      // 0..63
  const int part = tid & 3;            // K-quarter: 256 elems each

  for (int t = 0; t < SEQ; ++t) {
    // Pull current h (written last step into h_buf[t&1]) into LDS.
    const float* hcur = h_buf + (t & 1) * HID;
    for (int j = tid; j < HID; j += 256) h_lds[j] = hcur[j];
    __syncthreads();

    // 64 dot products, 4 threads each over 256 elements.
    {
      const float* wrow = w_lds + row_local * HID + part * 256;
      const float* hseg = h_lds + part * 256;
      float s = 0.f;
      #pragma unroll 8
      for (int k = 0; k < 256; ++k) s = fmaf(wrow[k], hseg[k], s);
      part_lds[tid] = s;
    }
    __syncthreads();

    if (tid < 64) {
      int gate = tid >> 4, u = tid & 15;
      int grow = gate * HID + u_base + u;
      float dot = part_lds[tid * 4] + part_lds[tid * 4 + 1] +
                  part_lds[tid * 4 + 2] + part_lds[tid * 4 + 3];
      gate_lds[tid] = xg[(long long)t * GATES + grow] + dot;
      if (t + 1 < SEQ)   // warm L2/L1 for next step's xg row (global_prefetch_b8)
        __builtin_prefetch(&xg[(long long)(t + 1) * GATES + grow], 0, 1);
    }
    __syncthreads();

    if (tid < UNITS_PER_BLOCK) {
      float gi = gate_lds[0 * 16 + tid];
      float gf = gate_lds[1 * 16 + tid];
      float gg = gate_lds[2 * 16 + tid];
      float go = gate_lds[3 * 16 + tid];
      gi = 1.f / (1.f + __expf(-gi));
      gf = 1.f / (1.f + __expf(-gf));
      gg = tanhf(gg);
      go = 1.f / (1.f + __expf(-go));
      float c = fmaf(gf, c_lds[tid], gi * gg);
      c_lds[tid] = c;
      h_buf[((t + 1) & 1) * HID + u_base + tid] = go * tanhf(c);
    }
    __threadfence();
    __syncthreads();

    // Grid barrier: monotone counter, target = (t+1)*RBLOCKS.
    if (tid == 0) {
      __hip_atomic_fetch_add(counter, 1u, __ATOMIC_ACQ_REL,
                             __HIP_MEMORY_SCOPE_AGENT);
      unsigned int target = (unsigned int)(t + 1) * RBLOCKS;
      while (__hip_atomic_load(counter, __ATOMIC_ACQUIRE,
                               __HIP_MEMORY_SCOPE_AGENT) < target) {
        __builtin_amdgcn_s_sleep(8);
      }
    }
    __syncthreads();
  }
}

// ---- K6: log_softmax over final h (h_buf[0], SEQ even) ---------------------
__global__ void log_softmax_kernel(const float* __restrict__ h,
                                   float* __restrict__ out) {
  __shared__ float red[256];
  int tid = threadIdx.x;
  float m = -INFINITY;
  for (int j = tid; j < HID; j += 256) m = fmaxf(m, h[j]);
  red[tid] = m;
  __syncthreads();
  for (int s = 128; s > 0; s >>= 1) {
    if (tid < s) red[tid] = fmaxf(red[tid], red[tid + s]);
    __syncthreads();
  }
  float mx = red[0];
  __syncthreads();
  float sum = 0.f;
  for (int j = tid; j < HID; j += 256) sum += expf(h[j] - mx);
  red[tid] = sum;
  __syncthreads();
  for (int s = 128; s > 0; s >>= 1) {
    if (tid < s) red[tid] += red[tid + s];
    __syncthreads();
  }
  float lse = logf(red[0]);
  for (int j = tid; j < HID; j += 256) out[j] = h[j] - mx - lse;
}

// ---------------------------------------------------------------------------
extern "C" void kernel_launch(void* const* d_in, const int* in_sizes, int n_in,
                              void* d_out, int out_size, void* d_ws, size_t ws_size,
                              hipStream_t stream) {
  (void)in_sizes; (void)n_in; (void)out_size; (void)ws_size;
  const int*   tokens = (const int*)  d_in[0];
  const float* emb    = (const float*)d_in[1];
  const float* W_ih   = (const float*)d_in[2];
  const float* W_hh   = (const float*)d_in[3];
  const float* b_ih   = (const float*)d_in[4];
  const float* b_hh   = (const float*)d_in[5];
  const float* h0     = (const float*)d_in[6];
  const float* c0     = (const float*)d_in[7];
  float* out = (float*)d_out;

  // Workspace layout (bytes):
  char* ws = (char*)d_ws;
  float*          xg      = (float*)(ws);                       // 64 MB
  unsigned short* Xb      = (unsigned short*)(ws + 67108864);   //  8 MB
  unsigned short* Wb      = (unsigned short*)(ws + 75497472);   //  8 MB
  float*          h_buf   = (float*)(ws + 83886080);            //  8 KB
  unsigned int*   counter = (unsigned int*)(ws + 83894272);     //  4 B

  embed_bf16_kernel<<<(SEQ * HID) / (256 * 4), 256, 0, stream>>>(tokens, emb, Xb);
  wih_bf16_kernel<<<(GATES * HID) / (256 * 4), 256, 0, stream>>>(W_ih, Wb);

  dim3 ggrid(SEQ / 64, GATES / 128);
  gemm_wmma_kernel<<<ggrid, 256, 0, stream>>>(
      (const __bf16*)Xb, (const __bf16*)Wb, b_ih, b_hh, xg);

  init_state_kernel<<<1, 1024, 0, stream>>>(h0, h_buf, counter);

  size_t lds_bytes = (64 * HID + HID + 256 + 64 + 16) * sizeof(float); // 267584
  lstm_scan_kernel<<<RBLOCKS, 256, lds_bytes, stream>>>(W_hh, xg, c0, h_buf,
                                                        counter);

  log_softmax_kernel<<<1, 256, 0, stream>>>(h_buf /* buffer 0 */, out);
}
